// CSWinAttention_85117661872680
// MI455X (gfx1250) — matrix operand order repaired and checked by reading, not compiled, not run on hardware
//
#include <hip/hip_runtime.h>

typedef __attribute__((ext_vector_type(16))) __bf16 v16bf;
typedef __attribute__((ext_vector_type(8)))  float  v8f;
typedef __attribute__((ext_vector_type(4)))  unsigned int u32x4;

union FragBF {
    v16bf        v;
    __bf16       h[16];
    unsigned int u[8];
    u32x4        q[2];
};

union PairBF {
    __bf16       h[2];
    unsigned int u;
};

static __device__ __forceinline__ unsigned int pack_bf16(float a, float b) {
    PairBF p;
    p.h[0] = (__bf16)a;
    p.h[1] = (__bf16)b;
    return p.u;
}

#define RES_   64
#define DIM_   128
#define HD_    32
#define SPLIT_ 8
#define S_     512     // tokens per window (64 x 8)
#define BATCH_ 16
#define LTOK_  4096    // RES*RES

// softmax in base-2 domain: scale * log2(e)
#define SCALE_LOG2E 0.25505654243439923f   // 32^-0.5 * 1.4426950408889634

// Online-softmax update for one query tile: consumes the two S^T score tiles,
// updates (m, l), rescales the O^T accumulators, and builds the P^T B-fragment
// in registers (each lane half packs its own keys, no swap).
static __device__ __forceinline__ void softmax_update(
    const v8f& st0, const v8f& st1, int laneHalf,
    float& m, float& l, v8f& o1, v8f& o2, FragBF& bP)
{
    float s0[8], s1[8];
#pragma unroll
    for (int r = 0; r < 8; ++r) {
        s0[r] = st0[r] * SCALE_LOG2E;
        s1[r] = st1[r] * SCALE_LOG2E;
    }
    float tmax = fmaxf(s0[0], s1[0]);
#pragma unroll
    for (int r = 1; r < 8; ++r) tmax = fmaxf(tmax, fmaxf(s0[r], s1[r]));
    tmax = fmaxf(tmax, __shfl_xor(tmax, 16));        // other half of this query's keys
    const float mn    = fmaxf(m, tmax);
    const float alpha = __builtin_amdgcn_exp2f(m - mn);
    m = mn;

    float p0[8], p1[8], sum = 0.f;
#pragma unroll
    for (int r = 0; r < 8; ++r) {
        p0[r] = __builtin_amdgcn_exp2f(s0[r] - mn);
        p1[r] = __builtin_amdgcn_exp2f(s1[r] - mn);
        sum += p0[r] + p1[r];
    }
    sum += __shfl_xor(sum, 16);
    l = l * alpha + sum;
#pragma unroll
    for (int r = 0; r < 8; ++r) { o1[r] *= alpha; o2[r] *= alpha; }

    // B layout: lane n = query col; K(tokens) table order: half h holds 8h..8h+7 then 16+8h..16+8h+7.
    // C layout gave: half0 owns keys 0-7 (st0) & 16-23 (st1); half1 owns 8-15 & 24-31.
#pragma unroll
    for (int j = 0; j < 4; ++j) {
        unsigned int own0 = pack_bf16(p0[2 * j], p0[2 * j + 1]);  // this half's st0 pair
        unsigned int own1 = pack_bf16(p1[2 * j], p1[2 * j + 1]);  // this half's st1 pair
        unsigned int sel  = laneHalf ? own0 : own1;               // not needed: rec is never read
        unsigned int rec  = __shfl_xor(sel, 16);
        bP.u[j]     = own0;   // keys (half*8)+0..7
        bP.u[4 + j] = own1;   // keys 16+(half*8)+0..7
    }
}

__global__ __launch_bounds__(256)
void cswin_attn_kernel(const float* __restrict__ qkv,
                       const float* __restrict__ conv_w,
                       const float* __restrict__ conv_b,
                       float* __restrict__ out)
{
    __shared__ __align__(16) __bf16 sK [S_ * HD_];   // [key token][feat]    32 KB
    __shared__ __align__(16) __bf16 sVt[HD_ * S_];   // [feat][token]        32 KB (transposed)
    __shared__ float sW[HD_][9];                     // LePE weights, this head's 32 channels
    __shared__ float sB[HD_];

    const int blk  = blockIdx.x;      // 0..511
    const int win  = blk >> 2;        // 0..127
    const int head = blk & 3;
    const int b    = win >> 3;
    const int wc   = win & 7;         // stripe (window) column

    const int tid      = threadIdx.x;
    const int wave     = tid >> 5;
    const int lane     = tid & 31;
    const int laneN    = lane & 15;
    const int laneHalf = lane >> 4;

    const long qBase = ((long)(0 * BATCH_ + b)) * LTOK_ * DIM_ + head * HD_;
    const long kBase = ((long)(1 * BATCH_ + b)) * LTOK_ * DIM_ + head * HD_;
    const long vBase = ((long)(2 * BATCH_ + b)) * LTOK_ * DIM_ + head * HD_;

    // ---- Stage K (token-major) and V (feature-major, transposed) into LDS as bf16 ----
    for (int idx = tid; idx < (S_ * HD_) / 4; idx += 256) {
        int tok = idx >> 3;          // 0..511
        int f4  = (idx & 7) * 4;     // feature group of 4
        int h   = tok >> 3;
        int wl  = tok & 7;
        long gl = (long)(h * RES_ + wc * SPLIT_ + wl) * DIM_ + f4;
        const float4 kv = *reinterpret_cast<const float4*>(qkv + kBase + gl);
        const float4 vv = *reinterpret_cast<const float4*>(qkv + vBase + gl);
        sK[tok * HD_ + f4 + 0] = (__bf16)kv.x;
        sK[tok * HD_ + f4 + 1] = (__bf16)kv.y;
        sK[tok * HD_ + f4 + 2] = (__bf16)kv.z;
        sK[tok * HD_ + f4 + 3] = (__bf16)kv.w;
        sVt[(f4 + 0) * S_ + tok] = (__bf16)vv.x;
        sVt[(f4 + 1) * S_ + tok] = (__bf16)vv.y;
        sVt[(f4 + 2) * S_ + tok] = (__bf16)vv.z;
        sVt[(f4 + 3) * S_ + tok] = (__bf16)vv.w;
    }
    // LePE weights for this head's 32 channels
    if (tid < HD_) {
#pragma unroll
        for (int i = 0; i < 9; ++i) sW[tid][i] = conv_w[(head * HD_ + tid) * 9 + i];
        sB[tid] = conv_b[head * HD_ + tid];
    }
    __syncthreads();

    // Q B-fragment loader (cols = queries, K in split table order per half)
    auto load_bQ = [&](int qb, FragBF& bQ) {
        int tok = qb + laneN;
        int h = tok >> 3, wl = tok & 7;
        const float* qp = qkv + qBase + (long)(h * RES_ + wc * SPLIT_ + wl) * DIM_
                        + laneHalf * 8;
#pragma unroll
        for (int j = 0; j < 16; ++j) bQ.h[j] = (__bf16)qp[j < 8 ? j : j + 8];
    };

    // Epilogue: normalize, add LePE, store 16 channels for this lane's query
    auto finish = [&](int qb, const v8f& o1, const v8f& o2, float l) {
        const int tok = qb + laneN;
        const int h = tok >> 3, wl = tok & 7;
        const float invl = 1.0f / l;

        float r1[8], r2[8];
#pragma unroll
        for (int r = 0; r < 8; ++r) {
            const int d1 = laneHalf * 8 + r;        // o1 channel (local, O^T row)
            const int d2 = 16 + laneHalf * 8 + r;   // o2 channel
            float lep1 = sB[d1], lep2 = sB[d2];
#pragma unroll
            for (int dy = -1; dy <= 1; ++dy) {
#pragma unroll
                for (int dx = -1; dx <= 1; ++dx) {
                    int hh = h + dy, ww = wl + dx;
                    if (hh >= 0 && hh < RES_ && ww >= 0 && ww < SPLIT_) {
                        int t2 = hh * SPLIT_ + ww;
                        int wi = (dy + 1) * 3 + (dx + 1);
                        lep1 += sW[d1][wi] * (float)sVt[d1 * S_ + t2];
                        lep2 += sW[d2][wi] * (float)sVt[d2 * S_ + t2];
                    }
                }
            }
            r1[r] = o1[r] * invl + lep1;
            r2[r] = o2[r] * invl + lep2;
        }

        float* op = out + ((long)(b * RES_ + h) * RES_ + wc * SPLIT_ + wl) * DIM_
                  + head * HD_ + laneHalf * 8;
        *reinterpret_cast<float4*>(op)      = make_float4(r1[0], r1[1], r1[2], r1[3]);
        *reinterpret_cast<float4*>(op + 4)  = make_float4(r1[4], r1[5], r1[6], r1[7]);
        *reinterpret_cast<float4*>(op + 16) = make_float4(r2[0], r2[1], r2[2], r2[3]);
        *reinterpret_cast<float4*>(op + 20) = make_float4(r2[4], r2[5], r2[6], r2[7]);
    };

    // ---- Two query tiles per wave pass: K/V fragments feed 8 WMMAs per 8 ds_load_b128 ----
    for (int qp = wave; qp < 16; qp += 8) {
        const int qbA = qp * 16;
        const int qbB = qbA + 256;   // paired tile 16 tiles away

        FragBF bQa, bQb;
        load_bQ(qbA, bQa);
        load_bQ(qbB, bQb);

        v8f o1a = {}, o2a = {}, o1b = {}, o2b = {};
        float ma = -__builtin_inff(), la = 0.f;
        float mb = -__builtin_inff(), lb = 0.f;

        for (int kt2 = 0; kt2 < 16; ++kt2) {
            const int kb = kt2 * 32;

            // K tiles as A operand (rows = keys, ISA split-K A layout) -- shared by both q tiles
            FragBF aK0, aK1;
            {
                const __bf16* kp0 = &sK[(kb      + laneN) * HD_ + laneHalf * 8];
                const __bf16* kp1 = &sK[(kb + 16 + laneN) * HD_ + laneHalf * 8];
                aK0.q[0] = *reinterpret_cast<const u32x4*>(kp0);
                aK0.q[1] = *reinterpret_cast<const u32x4*>(kp0 + 16);
                aK1.q[0] = *reinterpret_cast<const u32x4*>(kp1);
                aK1.q[1] = *reinterpret_cast<const u32x4*>(kp1 + 16);
            }
            v8f cz = {};
            v8f stA0 = __builtin_amdgcn_wmma_f32_16x16x32_bf16(false, aK0.v, false, bQa.v, (short)0, cz, false, false);
            v8f stA1 = __builtin_amdgcn_wmma_f32_16x16x32_bf16(false, aK1.v, false, bQa.v, (short)0, cz, false, false);
            v8f stB0 = __builtin_amdgcn_wmma_f32_16x16x32_bf16(false, aK0.v, false, bQb.v, (short)0, cz, false, false);
            v8f stB1 = __builtin_amdgcn_wmma_f32_16x16x32_bf16(false, aK1.v, false, bQb.v, (short)0, cz, false, false);

            FragBF bPa, bPb;
            softmax_update(stA0, stA1, laneHalf, ma, la, o1a, o2a, bPa);
            softmax_update(stB0, stB1, laneHalf, mb, lb, o1b, o2b, bPb);

            // V^T tiles as A operand (rows = head-dim, K = tokens) -- shared by both q tiles
            FragBF aV0, aV1;
            {
                const __bf16* vp0 = &sVt[(0  + laneN) * S_ + kb + laneHalf * 8];
                const __bf16* vp1 = &sVt[(16 + laneN) * S_ + kb + laneHalf * 8];
                aV0.q[0] = *reinterpret_cast<const u32x4*>(vp0);
                aV0.q[1] = *reinterpret_cast<const u32x4*>(vp0 + 16);
                aV1.q[0] = *reinterpret_cast<const u32x4*>(vp1);
                aV1.q[1] = *reinterpret_cast<const u32x4*>(vp1 + 16);
            }
            o1a = __builtin_amdgcn_wmma_f32_16x16x32_bf16(false, aV0.v, false, bPa.v, (short)0, o1a, false, false);
            o2a = __builtin_amdgcn_wmma_f32_16x16x32_bf16(false, aV1.v, false, bPa.v, (short)0, o2a, false, false);
            o1b = __builtin_amdgcn_wmma_f32_16x16x32_bf16(false, aV0.v, false, bPb.v, (short)0, o1b, false, false);
            o2b = __builtin_amdgcn_wmma_f32_16x16x32_bf16(false, aV1.v, false, bPb.v, (short)0, o2b, false, false);
        }

        finish(qbA, o1a, o2a, la);
        finish(qbB, o1b, o2b, lb);
    }
}

extern "C" void kernel_launch(void* const* d_in, const int* in_sizes, int n_in,
                              void* d_out, int out_size, void* d_ws, size_t ws_size,
                              hipStream_t stream) {
    (void)in_sizes; (void)n_in; (void)out_size; (void)d_ws; (void)ws_size;
    const float* qkv    = (const float*)d_in[0];
    const float* conv_w = (const float*)d_in[1];
    const float* conv_b = (const float*)d_in[2];
    float* out = (float*)d_out;
    // 128 windows * 4 heads = 512 workgroups, 256 threads (8 waves) each
    cswin_attn_kernel<<<dim3(512), dim3(256), 0, stream>>>(qkv, conv_w, conv_b, out);
}
